// SafetyEnhancedVehicleCoordinationGNN_56195352100879
// MI455X (gfx1250) — compile-verified, matrix operand order checked by reading, not run on hardware
//
#include <hip/hip_runtime.h>
#include <hip/hip_bf16.h>

#define NN 100000
#define NE 3200000
#define HD 64

typedef _Float16 half_t;
typedef __attribute__((ext_vector_type(16))) _Float16 v16h;
typedef __attribute__((ext_vector_type(8)))  _Float16 v8h;
typedef __attribute__((ext_vector_type(2)))  _Float16 v2h;
typedef __attribute__((ext_vector_type(8)))  float    v8f;

// K index inside a 32-deep fragment for 16-bit WMMA operands (A and B share the
// pattern): lanes 0-15 hold K {kb+0..kb+7, kb+16..kb+23} with kb=0,
// lanes 16-31 the same with kb=8 (ISA 7.12.2, 16-bit A 16x32 layout).
__device__ __forceinline__ int frag_k(int i, int lane) {
  int kb = (lane >> 4) << 3;
  return kb + (i < 8 ? i : i + 8);
}

// ---------------- weight packing: f32 [K x M] row-major -> f16 B-fragments ---
__global__ void pack_w_kernel(const float* __restrict__ W, half_t* __restrict__ P,
                              int K, int Kp, int M) {
  int t = blockIdx.x * blockDim.x + threadIdx.x;
  int tilesN = M >> 4;
  int nTiles = (Kp >> 5) * tilesN;
  int tile = t >> 5;
  if (tile >= nTiles) return;
  int lane = t & 31;
  int kt = tile / tilesN;
  int nt = tile - kt * tilesN;
  int n = (nt << 4) + (lane & 15);
  half_t* dstp = P + (((size_t)tile << 5) + lane) * 16;
#pragma unroll
  for (int i = 0; i < 16; ++i) {
    int k = (kt << 5) + frag_k(i, lane);
    float v = (k < K) ? W[(size_t)k * M + n] : 0.0f;
    dstp[i] = (half_t)v;
  }
}

// ---------------- x (N x 12 f32) -> x16 (N x 32 f16, zero padded) ------------
__global__ void cvt_x_kernel(const float* __restrict__ x, half_t* __restrict__ x16) {
  int t = blockIdx.x * blockDim.x + threadIdx.x;
  if (t >= NN * 32) return;
  int row = t >> 5, c = t & 31;
  x16[t] = (c < 12) ? (half_t)x[row * 12 + c] : (half_t)0.0f;
}

// ---------------- WMMA GEMM: C = act(A[N,KP]f16 @ Bpack + bias) --------------
// B-fragments + bias staged once per workgroup in LDS; 8 waves x one 16-row tile.
template <int KP, int MM, bool RELU, bool BIAS, bool OUTF, bool OUTH>
__global__ void wmma_gemm_kernel(const half_t* __restrict__ A,
                                 const half_t* __restrict__ Bp,
                                 const float* __restrict__ bias,
                                 float* __restrict__ Cf,
                                 half_t* __restrict__ Ch) {
  constexpr int TK = KP / 32;
  constexpr int TN = MM / 16;
  constexpr int BH = KP * MM;         // halves in the packed B block (<= 4096)
  constexpr int ROW_TILES = NN / 16;  // 6250, exact

  __shared__ __align__(32) half_t sB[BH];
  __shared__ float sBias[MM];
  for (int i = threadIdx.x; i < BH / 8; i += blockDim.x)
    ((v8h*)sB)[i] = ((const v8h*)Bp)[i];
  if (BIAS)
    for (int i = threadIdx.x; i < MM; i += blockDim.x) sBias[i] = bias[i];
  __syncthreads();

  int lane = threadIdx.x & 31;
  int tile = blockIdx.x * (blockDim.x >> 5) + (threadIdx.x >> 5);
  if (tile >= ROW_TILES) return;  // wave-uniform: EXEC stays all-ones
  int row = (tile << 4) + (lane & 15);
  int kb = (lane >> 4) << 3;

  v8f acc[TN] = {};
#pragma unroll
  for (int kt = 0; kt < TK; ++kt) {
    union { v16h v; v8h h[2]; } a;
    const half_t* ap = A + (size_t)row * KP + (kt << 5) + kb;
    a.h[0] = *(const v8h*)(ap);
    a.h[1] = *(const v8h*)(ap + 16);
#pragma unroll
    for (int nt = 0; nt < TN; ++nt) {
      v16h b = *(const v16h*)(sB + (((kt * TN + nt) << 5) + lane) * 16);
      acc[nt] = __builtin_amdgcn_wmma_f32_16x16x32_f16(
          false, a.v, false, b, (short)0, acc[nt], false, false);
    }
  }
  int rbase = (tile << 4) + ((lane >> 4) << 3);
  int cb = lane & 15;
#pragma unroll
  for (int nt = 0; nt < TN; ++nt) {
    int col = (nt << 4) + cb;
    float bv = BIAS ? sBias[col] : 0.0f;
#pragma unroll
    for (int r = 0; r < 8; ++r) {
      float v = acc[nt][r] + bv;
      if (RELU) v = fmaxf(v, 0.0f);
      size_t idx = (size_t)(rbase + r) * MM + col;
      if (OUTF) Cf[idx] = v;
      if (OUTH) Ch[idx] = (half_t)v;
    }
  }
}

// ---------------- degree / rsqrt --------------------------------------------
__global__ void deg_init_kernel(float* __restrict__ deg) {
  int i = blockIdx.x * blockDim.x + threadIdx.x;
  if (i < NN) deg[i] = 1.0f;  // self loop
}
__global__ void deg_count_kernel(const int* __restrict__ dst, float* __restrict__ deg) {
  int e = blockIdx.x * blockDim.x + threadIdx.x;
  if (e < NE) unsafeAtomicAdd(&deg[dst[e]], 1.0f);
}
__global__ void dinv_kernel(const float* __restrict__ deg, float* __restrict__ dinv) {
  int i = blockIdx.x * blockDim.x + threadIdx.x;
  if (i < NN) dinv[i] = rsqrtf(deg[i]);
}

// ---------------- agg = hW * dinv^2 + b (self loop + bias) -------------------
__global__ void self_init_kernel(const float* __restrict__ hW,
                                 const float* __restrict__ dinv,
                                 const float* __restrict__ b,
                                 float* __restrict__ agg) {
  int t = blockIdx.x * blockDim.x + threadIdx.x;
  if (t >= NN * HD) return;
  int i = t >> 6, k = t & 63;
  float di = dinv[i];
  agg[t] = hW[t] * di * di + b[k];
}

// ---------------- edge scatter: agg[dst] += hW[src] * dinv[s]*dinv[d] --------
// one wave per edge; indices/dinv are wave-uniform -> scalar loads; source row
// gathered as f16 (128B/edge), accumulation via f32 HW atomics.
__global__ void edge_scatter_kernel(const int* __restrict__ src,
                                    const int* __restrict__ dst,
                                    const float* __restrict__ dinv,
                                    const half_t* __restrict__ hW16,
                                    float* __restrict__ agg) {
  int lane = threadIdx.x & 31;
  int e = __builtin_amdgcn_readfirstlane(
      blockIdx.x * (blockDim.x >> 5) + (threadIdx.x >> 5));
  int s = src[e], d = dst[e];
  float norm = dinv[s] * dinv[d];
  const half_t* hs = hW16 + (size_t)s * HD;
  float* ad = agg + (size_t)d * HD;
  int c = lane * 2;
  v2h hv = *(const v2h*)(hs + c);
  unsafeAtomicAdd(&ad[c], (float)hv[0] * norm);
  unsafeAtomicAdd(&ad[c + 1], (float)hv[1] * norm);
}

// ---------------- ReLU + f16 downconvert ------------------------------------
__global__ void relu_h16_kernel(const float* __restrict__ agg, half_t* __restrict__ h16) {
  int t = blockIdx.x * blockDim.x + threadIdx.x;
  if (t >= NN * HD) return;
  float v = agg[t];
  h16[t] = (half_t)(v > 0.0f ? v : 0.0f);
}

// ---------------- head layer-2: out = act(r1 @ w + b), C = 32 ----------------
__global__ void head_out1_kernel(const half_t* __restrict__ R,
                                 const float* __restrict__ w,
                                 const float* __restrict__ b, int act,
                                 float* __restrict__ out) {
  int n = blockIdx.x * blockDim.x + threadIdx.x;
  if (n >= NN) return;
  const half_t* r = R + (size_t)n * 32;
  float s = b[0];
#pragma unroll
  for (int j = 0; j < 32; ++j) s += (float)r[j] * w[j];
  out[n] = (act == 0) ? tanhf(s) : (1.0f / (1.0f + expf(-s)));
}

// ---------------- route head: 32 -> 3 + softmax ------------------------------
__global__ void route_out_kernel(const half_t* __restrict__ R,
                                 const float* __restrict__ W2,
                                 const float* __restrict__ b2,
                                 float* __restrict__ out) {
  int n = blockIdx.x * blockDim.x + threadIdx.x;
  if (n >= NN) return;
  const half_t* r = R + (size_t)n * 32;
  float z[3] = {b2[0], b2[1], b2[2]};
#pragma unroll
  for (int i = 0; i < 32; ++i) {
    float ri = (float)r[i];
    z[0] += ri * W2[i * 3 + 0];
    z[1] += ri * W2[i * 3 + 1];
    z[2] += ri * W2[i * 3 + 2];
  }
  float m = fmaxf(z[0], fmaxf(z[1], z[2]));
  float e0 = expf(z[0] - m), e1 = expf(z[1] - m), e2 = expf(z[2] - m);
  float inv = 1.0f / (e0 + e1 + e2);
  out[(size_t)n * 3 + 0] = e0 * inv;
  out[(size_t)n * 3 + 1] = e1 * inv;
  out[(size_t)n * 3 + 2] = e2 * inv;
}

extern "C" void kernel_launch(void* const* d_in, const int* in_sizes, int n_in,
                              void* d_out, int out_size, void* d_ws, size_t ws_size,
                              hipStream_t stream) {
  (void)in_sizes; (void)n_in; (void)out_size; (void)ws_size;
  // ---- inputs, in setup_inputs() insertion order (recursive) ----
  const float* x      = (const float*)d_in[0];
  const int*   ei     = (const int*)d_in[1];
  const int*   srcIdx = ei;
  const int*   dstIdx = ei + NE;
  const float* encW = (const float*)d_in[2];
  const float* encB = (const float*)d_in[3];
  const float* gcnW[3] = {(const float*)d_in[4], (const float*)d_in[6], (const float*)d_in[8]};
  const float* gcnB[3] = {(const float*)d_in[5], (const float*)d_in[7], (const float*)d_in[9]};
  const float* priW1 = (const float*)d_in[10], *priB1 = (const float*)d_in[11];
  const float* priW2 = (const float*)d_in[12], *priB2 = (const float*)d_in[13];
  const float* cooW1 = (const float*)d_in[14], *cooB1 = (const float*)d_in[15];
  const float* cooW2 = (const float*)d_in[16], *cooB2 = (const float*)d_in[17];
  const float* urgW1 = (const float*)d_in[18], *urgB1 = (const float*)d_in[19];
  const float* urgW2 = (const float*)d_in[20], *urgB2 = (const float*)d_in[21];
  const float* safW1 = (const float*)d_in[22], *safB1 = (const float*)d_in[23];
  const float* safW2 = (const float*)d_in[24], *safB2 = (const float*)d_in[25];
  const float* safW3 = (const float*)d_in[26], *safB3 = (const float*)d_in[27];
  const float* spdW1 = (const float*)d_in[28], *spdB1 = (const float*)d_in[29];
  const float* spdW2 = (const float*)d_in[30], *spdB2 = (const float*)d_in[31];
  const float* rteW1 = (const float*)d_in[32], *rteB1 = (const float*)d_in[33];
  const float* rteW2 = (const float*)d_in[34], *rteB2 = (const float*)d_in[35];
  float* out = (float*)d_out;

  // ---- workspace carve-out (256B aligned) ----
  char* ws = (char*)d_ws;
  size_t off = 0;
  auto take = [&](size_t bytes) -> void* {
    void* p = (void*)(ws + off);
    off += (bytes + 255) & ~(size_t)255;
    return p;
  };
  half_t* x16  = (half_t*)take((size_t)NN * 32 * 2);
  half_t* h16  = (half_t*)take((size_t)NN * 64 * 2);
  float*  hW   = (float*) take((size_t)NN * 64 * 4);
  half_t* hW16 = (half_t*)take((size_t)NN * 64 * 2);
  float*  agg  = (float*) take((size_t)NN * 64 * 4);
  float*  deg  = (float*) take((size_t)NN * 4);
  float*  dinv = (float*) take((size_t)NN * 4);
  half_t* r1   = (half_t*)take((size_t)NN * 64 * 2);
  half_t* r2   = (half_t*)take((size_t)NN * 32 * 2);
  half_t* encP = (half_t*)take((size_t)32 * 64 * 2);
  half_t* gcnP[3];
  for (int l = 0; l < 3; ++l) gcnP[l] = (half_t*)take((size_t)64 * 64 * 2);
  half_t* priP  = (half_t*)take((size_t)64 * 32 * 2);
  half_t* cooP  = (half_t*)take((size_t)64 * 32 * 2);
  half_t* urgP  = (half_t*)take((size_t)64 * 32 * 2);
  half_t* spdP  = (half_t*)take((size_t)64 * 32 * 2);
  half_t* rteP  = (half_t*)take((size_t)64 * 32 * 2);
  half_t* safP1 = (half_t*)take((size_t)64 * 64 * 2);
  half_t* safP2 = (half_t*)take((size_t)64 * 32 * 2);

  auto pack = [&](const float* W, half_t* P, int K, int Kp, int M) {
    int threads = (Kp / 32) * (M / 16) * 32;
    pack_w_kernel<<<dim3((threads + 127) / 128), dim3(128), 0, stream>>>(W, P, K, Kp, M);
  };
  pack(encW, encP, 12, 32, 64);
  for (int l = 0; l < 3; ++l) pack(gcnW[l], gcnP[l], 64, 64, 64);
  pack(priW1, priP, 64, 64, 32);
  pack(cooW1, cooP, 64, 64, 32);
  pack(urgW1, urgP, 64, 64, 32);
  pack(spdW1, spdP, 64, 64, 32);
  pack(rteW1, rteP, 64, 64, 32);
  pack(safW1, safP1, 64, 64, 64);
  pack(safW2, safP2, 64, 64, 32);

  const int GEMM_BLOCKS = (NN / 16 + 7) / 8;  // 8 waves per 256-thread block

  // ---- encoder: h = relu(x @ We + be) ----
  cvt_x_kernel<<<dim3((NN * 32 + 255) / 256), dim3(256), 0, stream>>>(x, x16);
  wmma_gemm_kernel<32, 64, true, true, false, true>
      <<<dim3(GEMM_BLOCKS), dim3(256), 0, stream>>>(x16, encP, encB, nullptr, h16);

  // ---- degree normalization (dst fixed across layers: compute once) ----
  deg_init_kernel<<<dim3((NN + 255) / 256), dim3(256), 0, stream>>>(deg);
  deg_count_kernel<<<dim3((NE + 255) / 256), dim3(256), 0, stream>>>(dstIdx, deg);
  dinv_kernel<<<dim3((NN + 255) / 256), dim3(256), 0, stream>>>(deg, dinv);

  // ---- 3 GCN layers ----
  for (int l = 0; l < 3; ++l) {
    wmma_gemm_kernel<64, 64, false, false, true, true>
        <<<dim3(GEMM_BLOCKS), dim3(256), 0, stream>>>(h16, gcnP[l], nullptr, hW, hW16);
    self_init_kernel<<<dim3((NN * 64 + 255) / 256), dim3(256), 0, stream>>>(
        hW, dinv, gcnB[l], agg);
    edge_scatter_kernel<<<dim3(NE / 8), dim3(256), 0, stream>>>(
        srcIdx, dstIdx, dinv, hW16, agg);
    relu_h16_kernel<<<dim3((NN * 64 + 255) / 256), dim3(256), 0, stream>>>(agg, h16);
  }

  const int HB = (NN + 255) / 256;
  // ---- priority (tanh) ----
  wmma_gemm_kernel<64, 32, true, true, false, true>
      <<<dim3(GEMM_BLOCKS), dim3(256), 0, stream>>>(h16, priP, priB1, nullptr, r1);
  head_out1_kernel<<<dim3(HB), dim3(256), 0, stream>>>(r1, priW2, priB2, 0, out + 0);
  // ---- cooperation (sigmoid) ----
  wmma_gemm_kernel<64, 32, true, true, false, true>
      <<<dim3(GEMM_BLOCKS), dim3(256), 0, stream>>>(h16, cooP, cooB1, nullptr, r1);
  head_out1_kernel<<<dim3(HB), dim3(256), 0, stream>>>(r1, cooW2, cooB2, 1, out + NN);
  // ---- urgency (sigmoid) ----
  wmma_gemm_kernel<64, 32, true, true, false, true>
      <<<dim3(GEMM_BLOCKS), dim3(256), 0, stream>>>(h16, urgP, urgB1, nullptr, r1);
  head_out1_kernel<<<dim3(HB), dim3(256), 0, stream>>>(r1, urgW2, urgB2, 1, out + 2 * NN);
  // ---- safety (64 -> 64 -> 32 -> 1, sigmoid) ----
  wmma_gemm_kernel<64, 64, true, true, false, true>
      <<<dim3(GEMM_BLOCKS), dim3(256), 0, stream>>>(h16, safP1, safB1, nullptr, r1);
  wmma_gemm_kernel<64, 32, true, true, false, true>
      <<<dim3(GEMM_BLOCKS), dim3(256), 0, stream>>>(r1, safP2, safB2, nullptr, r2);
  head_out1_kernel<<<dim3(HB), dim3(256), 0, stream>>>(r2, safW3, safB3, 1, out + 3 * NN);
  // ---- speed_adjustment (tanh) ----
  wmma_gemm_kernel<64, 32, true, true, false, true>
      <<<dim3(GEMM_BLOCKS), dim3(256), 0, stream>>>(h16, spdP, spdB1, nullptr, r1);
  head_out1_kernel<<<dim3(HB), dim3(256), 0, stream>>>(r1, spdW2, spdB2, 0, out + 4 * NN);
  // ---- route_preference (softmax over 3) ----
  wmma_gemm_kernel<64, 32, true, true, false, true>
      <<<dim3(GEMM_BLOCKS), dim3(256), 0, stream>>>(h16, rteP, rteB1, nullptr, r1);
  route_out_kernel<<<dim3(HB), dim3(256), 0, stream>>>(r1, rteW2, rteB2, out + 5 * NN);
}
